// CrossModalTemporalAligner_82755429859456
// MI455X (gfx1250) — matrix operand order repaired and checked by reading, not compiled, not run on hardware
//
#include <hip/hip_runtime.h>
#include <hip/hip_bf16.h>

typedef __bf16 bf16_t;
typedef __attribute__((ext_vector_type(16))) __bf16 v16bf;
typedef __attribute__((ext_vector_type(8)))  __bf16 v8bf;
typedef __attribute__((ext_vector_type(8)))  float   v8f;

// Problem constants (reference shapes)
static constexpr int kB  = 4;
static constexpr int kTi = 512;
static constexpr int kTj = 512;
static constexpr int kN  = 64;
static constexpr int kD  = 512;
static constexpr int kT  = 512;   // Ti == Tj

static __device__ inline v8f wmma_bf16(v16bf a, v16bf b, v8f c) {
  // D = A(16x32 bf16) x B(32x16 bf16) + C(16x16 f32)
  return __builtin_amdgcn_wmma_f32_16x16x32_bf16(
      false, a, false, b, (short)0, c, false, false);
}

// A-operand fragment from an f32 row: row = lane&15, K-chunks by lane>>4.
// elements 0..7  = K kc..kc+7 ; elements 8..15 = K kc+8..kc+15
static __device__ inline v16bf load_a_f32(const float* __restrict__ row, int kc) {
  float4 f0 = *(const float4*)(row + kc);
  float4 f1 = *(const float4*)(row + kc + 4);
  float4 f2 = *(const float4*)(row + kc + 8);
  float4 f3 = *(const float4*)(row + kc + 12);
  v16bf a;
  a[0]=(bf16_t)f0.x; a[1]=(bf16_t)f0.y; a[2]=(bf16_t)f0.z; a[3]=(bf16_t)f0.w;
  a[4]=(bf16_t)f1.x; a[5]=(bf16_t)f1.y; a[6]=(bf16_t)f1.z; a[7]=(bf16_t)f1.w;
  a[8]=(bf16_t)f2.x; a[9]=(bf16_t)f2.y; a[10]=(bf16_t)f2.z; a[11]=(bf16_t)f2.w;
  a[12]=(bf16_t)f3.x; a[13]=(bf16_t)f3.y; a[14]=(bf16_t)f3.z; a[15]=(bf16_t)f3.w;
  return a;
}

// A-operand fragment from bf16 LDS row.
static __device__ inline v16bf load_a_lds(const bf16_t* rowp, int kc) {
  v8bf a0 = *(const v8bf*)(rowp + kc);
  v8bf a1 = *(const v8bf*)(rowp + kc + 8);
  v16bf a;
#pragma unroll
  for (int i = 0; i < 8; ++i) { a[i] = a0[i]; a[i + 8] = a1[i]; }
  return a;
}

// -----------------------------------------------------------------------------
// Kernel 0: one-shot weight conversion f32 -> bf16 (3 x 512 x 512 elements).
// Removes all B-fragment conversion VALU from the projection GEMM and halves
// W read bandwidth there.
// -----------------------------------------------------------------------------
__global__ __launch_bounds__(256) void cmta_cvtw_kernel(
    const float* __restrict__ Wq, const float* __restrict__ Wk,
    const float* __restrict__ Wv,
    bf16_t* __restrict__ Wqb, bf16_t* __restrict__ Wkb,
    bf16_t* __restrict__ Wvb)
{
  const int z = blockIdx.z;
  const float* __restrict__ src = (z == 0) ? Wq : ((z == 1) ? Wk : Wv);
  bf16_t* __restrict__ dst      = (z == 0) ? Wqb : ((z == 1) ? Wkb : Wvb);
  const int i = (blockIdx.x * 256 + threadIdx.x) * 8;
  float4 a = *(const float4*)(src + i);
  float4 b = *(const float4*)(src + i + 4);
  v8bf p;
  p[0]=(bf16_t)a.x; p[1]=(bf16_t)a.y; p[2]=(bf16_t)a.z; p[3]=(bf16_t)a.w;
  p[4]=(bf16_t)b.x; p[5]=(bf16_t)b.y; p[6]=(bf16_t)b.z; p[7]=(bf16_t)b.w;
  *(v8bf*)(dst + i) = p;
}

// -----------------------------------------------------------------------------
// Kernel 1: projections.  y = x @ W^T + b  (NT GEMM), f32 activations,
// bf16 weights (pre-converted), bf16 out.
//   z==0: Q from H_i -> Qb[b][n][t][d]
//   z==1: K AND V from H_j (fused: one read of H_j)
//         K -> Kb[b][n][s][d] ; V -> Vt[b][n][d][s] (transposed)
// M-dim ordering: m = ((b*kN + n)*kT + t)  (t fastest).
// Block: 256 threads = 8 waves; wave tile 16 rows x 128 cols; block 32 x 512.
// -----------------------------------------------------------------------------
__global__ __launch_bounds__(256) void cmta_proj_kernel(
    const float* __restrict__ Hi, const float* __restrict__ Hj,
    const bf16_t* __restrict__ Wqb, const float* __restrict__ bq,
    const bf16_t* __restrict__ Wkb, const float* __restrict__ bk,
    const bf16_t* __restrict__ Wvb, const float* __restrict__ bv,
    bf16_t* __restrict__ Qb, bf16_t* __restrict__ Kb, bf16_t* __restrict__ Vt)
{
  const int z    = blockIdx.z;            // 0: Q ; 1: K+V
  const int lane = threadIdx.x & 31;
  const int wave = threadIdx.x >> 5;
  const int lo   = lane & 15;
  const int hi   = lane >> 4;

  const int rowBase = blockIdx.x * 32 + (wave >> 2) * 16;   // 16-row tile
  const int colWave = (wave & 3) * 128;                     // 128-col strip

  const int b  = rowBase / (kN * kT);
  const int n  = (rowBase / kT) % kN;
  const int t0 = rowBase % kT;

  const float* __restrict__ X = (z == 0) ? Hi : Hj;
  const float* __restrict__ xrow =
      X + (((size_t)b * kT + (t0 + lo)) * kN + n) * kD;

  const size_t headBase = (size_t)(b * kN + n);

  if (z == 0) {
    v8f acc[8] = {};
    for (int k = 0; k < kD; k += 32) {
      __builtin_prefetch(xrow + k + 128, 0, 0);
      v16bf a = load_a_f32(xrow, k + hi * 16);
      const int ks = k + hi * 16;
#pragma unroll
      for (int c = 0; c < 8; ++c) {
        const int col = colWave + c * 16 + lo;
        v16bf bf = *(const v16bf*)(Wqb + (size_t)col * kD + ks);
        acc[c] = wmma_bf16(a, bf, acc[c]);
      }
    }
#pragma unroll
    for (int c = 0; c < 8; ++c) {
      const int col = colWave + c * 16 + lo;
      const float bb = bq[col];
#pragma unroll
      for (int r = 0; r < 8; ++r) {
        const int tt = t0 + hi * 8 + r;
        Qb[(headBase * kT + tt) * kD + col] = (bf16_t)(acc[c][r] + bb);
      }
    }
  } else {
    v8f accK[8] = {};
    v8f accV[8] = {};
    for (int k = 0; k < kD; k += 32) {
      __builtin_prefetch(xrow + k + 128, 0, 0);
      v16bf a = load_a_f32(xrow, k + hi * 16);
      const int ks = k + hi * 16;
#pragma unroll
      for (int c = 0; c < 8; ++c) {
        const int col = colWave + c * 16 + lo;
        v16bf bfk = *(const v16bf*)(Wkb + (size_t)col * kD + ks);
        accK[c] = wmma_bf16(a, bfk, accK[c]);
        v16bf bfv = *(const v16bf*)(Wvb + (size_t)col * kD + ks);
        accV[c] = wmma_bf16(a, bfv, accV[c]);
      }
    }
#pragma unroll
    for (int c = 0; c < 8; ++c) {
      const int col = colWave + c * 16 + lo;
      const float bbk = bk[col];
      const float bbv = bv[col];
      // K: row-major [s][d]
#pragma unroll
      for (int r = 0; r < 8; ++r) {
        const int ss = t0 + hi * 8 + r;
        Kb[(headBase * kT + ss) * kD + col] = (bf16_t)(accK[c][r] + bbk);
      }
      // V: transposed [d][s] -> 8 contiguous bf16 per lane
      v8bf pk;
#pragma unroll
      for (int r = 0; r < 8; ++r) pk[r] = (bf16_t)(accV[c][r] + bbv);
      *(v8bf*)(Vt + (headBase * kD + col) * kT + t0 + hi * 8) = pk;
    }
  }
}

// -----------------------------------------------------------------------------
// Kernel 2: fused attention per (b, n, 32-row Ti tile).
// 8 waves; wave w owns Tj columns [64w,64w+64) for scores and d columns
// [64w,64w+64) for the output GEMM. Two 16-row sub-tiles (u=0,1) share every
// B-fragment, doubling arithmetic intensity on K/Vt reads from L2.
// -----------------------------------------------------------------------------
__global__ __launch_bounds__(256) void cmta_attn_kernel(
    const bf16_t* __restrict__ Qb, const bf16_t* __restrict__ Kb,
    const bf16_t* __restrict__ Vt,
    const float* __restrict__ log_gamma, const float* __restrict__ log_tau,
    float* __restrict__ out)
{
  __shared__ bf16_t Qs[32 * kD];
  __shared__ bf16_t Ps[32 * kTj];
  __shared__ float  redM[8][32];
  __shared__ float  redS[8][32];

  const int tTiles = kTi / 32;
  const int idx = blockIdx.x;
  const int tt = idx % tTiles;
  const int n  = (idx / tTiles) % kN;
  const int b  = idx / (tTiles * kN);
  const int t0 = tt * 32;

  const int lane = threadIdx.x & 31;
  const int wave = threadIdx.x >> 5;
  const int lo   = lane & 15;
  const int hi   = lane >> 4;

  const float tau   = fmaxf(__expf(log_tau[0]),   0.01f);
  const float gamma = fmaxf(__expf(log_gamma[0]), 0.01f);
  const float scale = 1.0f / (sqrtf((float)kD) * tau);
  const float invTi = 1.0f / (float)(kTi - 1);
  const float invTj = 1.0f / (float)(kTj - 1);

  const size_t headBase = (size_t)(b * kN + n);

  // ---- stage Q tile (32 x 512 bf16) into LDS, coalesced ----
  {
    const int r = threadIdx.x >> 3;          // 0..31
    const int c = (threadIdx.x & 7) * 64;    // 0..448 step 64
    const bf16_t* src = Qb + ((headBase * kTi) + t0 + r) * kD + c;
    bf16_t* dst = &Qs[r * kD + c];
#pragma unroll
    for (int q = 0; q < 4; ++q)
      *(v16bf*)(dst + q * 16) = *(const v16bf*)(src + q * 16);
  }
  __syncthreads();

  // ---- phase 1: S = Q K^T over this wave's 64 Tj columns ----
  const int sWave = wave * 64;
  v8f acc[2][4] = {};
  for (int k = 0; k < kD; k += 32) {
    const int kc = k + hi * 16;
    v16bf a0 = load_a_lds(&Qs[(lo)      * kD], kc);
    v16bf a1 = load_a_lds(&Qs[(16 + lo) * kD], kc);
    const int ks = k + hi * 16;
#pragma unroll
    for (int j = 0; j < 4; ++j) {
      const int s = sWave + j * 16 + lo;
      v16bf bf = *(const v16bf*)(Kb + (headBase * kTj + s) * kD + ks);
      acc[0][j] = wmma_bf16(a0, bf, acc[0][j]);
      acc[1][j] = wmma_bf16(a1, bf, acc[1][j]);
    }
  }

  // ---- scale + temporal decay bias, row-max ----
  float vals[2][4][8];
  float m[2][8];
#pragma unroll
  for (int u = 0; u < 2; ++u)
#pragma unroll
    for (int r = 0; r < 8; ++r) m[u][r] = -3.0e30f;
#pragma unroll
  for (int j = 0; j < 4; ++j) {
    const int s = sWave + j * 16 + lo;
    const float tj = (float)s * invTj;
#pragma unroll
    for (int u = 0; u < 2; ++u)
#pragma unroll
      for (int r = 0; r < 8; ++r) {
        const int t = t0 + u * 16 + hi * 8 + r;
        const float dist = fabsf((float)t * invTi - tj);
        const float dl = __logf(__expf(-gamma * dist) + 1e-8f);
        const float v = acc[u][j][r] * scale + dl;
        vals[u][j][r] = v;
        m[u][r] = fmaxf(m[u][r], v);
      }
  }
#pragma unroll
  for (int off = 1; off < 16; off <<= 1)
#pragma unroll
    for (int u = 0; u < 2; ++u)
#pragma unroll
      for (int r = 0; r < 8; ++r)
        m[u][r] = fmaxf(m[u][r], __shfl_xor(m[u][r], off, 32));
  if (lo == 0) {
#pragma unroll
    for (int u = 0; u < 2; ++u)
#pragma unroll
      for (int r = 0; r < 8; ++r) redM[wave][u * 16 + hi * 8 + r] = m[u][r];
  }
  __syncthreads();
  float rowM[2][8];
#pragma unroll
  for (int u = 0; u < 2; ++u)
#pragma unroll
    for (int r = 0; r < 8; ++r) {
      const int row = u * 16 + hi * 8 + r;
      float mm = redM[0][row];
#pragma unroll
      for (int w = 1; w < 8; ++w) mm = fmaxf(mm, redM[w][row]);
      rowM[u][r] = mm;
    }

  // ---- exp + row-sum ----
  float sum[2][8];
#pragma unroll
  for (int u = 0; u < 2; ++u)
#pragma unroll
    for (int r = 0; r < 8; ++r) sum[u][r] = 0.0f;
#pragma unroll
  for (int u = 0; u < 2; ++u)
#pragma unroll
    for (int j = 0; j < 4; ++j)
#pragma unroll
      for (int r = 0; r < 8; ++r) {
        const float p = __expf(vals[u][j][r] - rowM[u][r]);
        vals[u][j][r] = p;
        sum[u][r] += p;
      }
#pragma unroll
  for (int off = 1; off < 16; off <<= 1)
#pragma unroll
    for (int u = 0; u < 2; ++u)
#pragma unroll
      for (int r = 0; r < 8; ++r)
        sum[u][r] += __shfl_xor(sum[u][r], off, 32);
  if (lo == 0) {
#pragma unroll
    for (int u = 0; u < 2; ++u)
#pragma unroll
      for (int r = 0; r < 8; ++r) redS[wave][u * 16 + hi * 8 + r] = sum[u][r];
  }
  __syncthreads();
  float inv[2][8];
#pragma unroll
  for (int u = 0; u < 2; ++u)
#pragma unroll
    for (int r = 0; r < 8; ++r) {
      const int row = u * 16 + hi * 8 + r;
      float ss = redS[0][row];
#pragma unroll
      for (int w = 1; w < 8; ++w) ss += redS[w][row];
      inv[u][r] = 1.0f / ss;
    }

  // ---- write normalized P (bf16) into LDS ----
#pragma unroll
  for (int u = 0; u < 2; ++u)
#pragma unroll
    for (int j = 0; j < 4; ++j)
#pragma unroll
      for (int r = 0; r < 8; ++r) {
        const int row = u * 16 + hi * 8 + r;
        Ps[row * kTj + sWave + j * 16 + lo] =
            (bf16_t)(vals[u][j][r] * inv[u][r]);
      }
  __syncthreads();

  // ---- phase 2: out = P @ V  (NT with Vt), this wave's 64 d columns ----
  const int dWave = wave * 64;
  v8f oacc[2][4] = {};
  for (int k = 0; k < kTj; k += 32) {
    const int kc = k + hi * 16;
    v16bf a0 = load_a_lds(&Ps[(lo)      * kTj], kc);
    v16bf a1 = load_a_lds(&Ps[(16 + lo) * kTj], kc);
    const int ks = k + hi * 16;
#pragma unroll
    for (int j = 0; j < 4; ++j) {
      const int d = dWave + j * 16 + lo;
      v16bf bf = *(const v16bf*)(Vt + (headBase * kD + d) * kTj + ks);
      oacc[0][j] = wmma_bf16(a0, bf, oacc[0][j]);
      oacc[1][j] = wmma_bf16(a1, bf, oacc[1][j]);
    }
  }

  // ---- store output: out[b][t][n][d], f32 ----
#pragma unroll
  for (int u = 0; u < 2; ++u)
#pragma unroll
    for (int j = 0; j < 4; ++j) {
      const int d = dWave + j * 16 + lo;
#pragma unroll
      for (int r = 0; r < 8; ++r) {
        const int t = t0 + u * 16 + hi * 8 + r;
        out[(((size_t)b * kTi + t) * kN + n) * kD + d] = oacc[u][j][r];
      }
    }
}

// -----------------------------------------------------------------------------
extern "C" void kernel_launch(void* const* d_in, const int* in_sizes, int n_in,
                              void* d_out, int out_size, void* d_ws, size_t ws_size,
                              hipStream_t stream) {
  const float* Hi        = (const float*)d_in[0];
  const float* Hj        = (const float*)d_in[1];
  const float* Wq        = (const float*)d_in[2];
  const float* bq        = (const float*)d_in[3];
  const float* Wk        = (const float*)d_in[4];
  const float* bk        = (const float*)d_in[5];
  const float* Wv        = (const float*)d_in[6];
  const float* bv        = (const float*)d_in[7];
  const float* log_gamma = (const float*)d_in[8];
  const float* log_tau   = (const float*)d_in[9];
  float* out             = (float*)d_out;

  const size_t matElems = (size_t)kB * kN * kT * kD;  // 67,108,864 bf16 each
  const size_t wElems   = (size_t)kD * kD;            // 262,144 bf16 each
  bf16_t* Qb  = (bf16_t*)d_ws;
  bf16_t* Kb  = Qb + matElems;
  bf16_t* Vt  = Kb + matElems;
  bf16_t* Wqb = Vt + matElems;
  bf16_t* Wkb = Wqb + wElems;
  bf16_t* Wvb = Wkb + wElems;

  // Weights f32 -> bf16 (once per call; 2048 elems per block * 128 blocks * 3)
  {
    dim3 grid((kD * kD) / (256 * 8), 1, 3);
    cmta_cvtw_kernel<<<grid, 256, 0, stream>>>(Wq, Wk, Wv, Wqb, Wkb, Wvb);
  }
  // Projections: M=131072 rows / 32 per block; z=0 -> Q, z=1 -> fused K+V
  {
    dim3 grid((kB * kN * kT) / 32, 1, 2);
    cmta_proj_kernel<<<grid, 256, 0, stream>>>(Hi, Hj, Wqb, bq, Wkb, bk,
                                               Wvb, bv, Qb, Kb, Vt);
  }
  // Attention: one block per (b, n, 32-row Ti tile)
  {
    dim3 grid(kB * kN * (kTi / 32));
    cmta_attn_kernel<<<grid, 256, 0, stream>>>(Qb, Kb, Vt, log_gamma, log_tau,
                                               out);
  }
}
